// WindowedAttention_84482006712579
// MI455X (gfx1250) — matrix operand order repaired and checked by reading, not compiled, not run on hardware
//
#include <hip/hip_runtime.h>
#include <hip/hip_bf16.h>

// ---------- types ----------
typedef __attribute__((ext_vector_type(16))) _Float16 v16h;
typedef __attribute__((ext_vector_type(8)))  float    v8f;
typedef __attribute__((ext_vector_type(4)))  float    v4f;
typedef __attribute__((ext_vector_type(4)))  unsigned int v4u;
typedef __attribute__((ext_vector_type(2)))  unsigned int v2u;

union Frag16 { v16h h; v4u u[2]; };          // 32B WMMA f16 operand
union H16   { v4u u[2]; _Float16 h[16]; };   // unpack helper
union H4    { v2u u; _Float16 h[4]; };
union F8    { v8f v; v4f q[2]; };            // C-operand assembly

#define RES 14
#define NTOK 196          // RES*RES
#define NHEAD 24
#define HDIM 32
#define DIM 768
#define NPAD 224          // 7 * 32  (token dim padded for WMMA K-steps)
#define MTILES 13         // 13 * 16 = 208 >= 196 query rows
#define MROWS 25088       // BATCH * NTOK
#define SCALE 0.17677669529663687f  // 1/sqrt(32)
#define NEGBIG -1e30f

// ---------- fp32 -> f16 convert ----------
__global__ void cvt_f16(const float* __restrict__ in, _Float16* __restrict__ out, int n4) {
  int i = blockIdx.x * 256 + threadIdx.x;
  if (i >= n4) return;
  v4f v = *(const v4f*)(in + 4 * (long)i);
  H4 o;
  o.h[0] = (_Float16)v[0]; o.h[1] = (_Float16)v[1];
  o.h[2] = (_Float16)v[2]; o.h[3] = (_Float16)v[3];
  *(v2u*)(out + 4 * (long)i) = o.u;
}

// qkv_w [768][2304]: convert to f16, pre-scaling the Q columns (<768) by 1/sqrt(32)
__global__ void cvt_f16_qkvw(const float* __restrict__ in, _Float16* __restrict__ out, int n4) {
  int i = blockIdx.x * 256 + threadIdx.x;
  if (i >= n4) return;
  long base = 4 * (long)i;
  float sc = ((int)(base % (3 * DIM)) < DIM) ? SCALE : 1.f;   // groups of 4 never straddle 768
  v4f v = *(const v4f*)(in + base);
  H4 o;
  o.h[0] = (_Float16)(v[0] * sc); o.h[1] = (_Float16)(v[1] * sc);
  o.h[2] = (_Float16)(v[2] * sc); o.h[3] = (_Float16)(v[3] * sc);
  *(v2u*)(out + base) = o.u;
}

// qkv_b: scaled copy (Q part * 1/sqrt(32))
__global__ void scale_qkv_b(const float* __restrict__ in, float* __restrict__ out) {
  int i = blockIdx.x * 256 + threadIdx.x;
  if (i < 3 * DIM) out[i] = in[i] * ((i < DIM) ? SCALE : 1.f);
}

// ---------- CPB: log-spaced coordinate ----------
__device__ __forceinline__ float cpb_coord(int i) {
  float c = (float)(i - (RES - 1)) / (float)(RES - 1) * 8.0f;
  float a = fabsf(c);
  float s = (c > 0.f) ? 1.f : ((c < 0.f) ? -1.f : 0.f);
  return s * __log2f(a + 1.f) * (1.0f / 3.0f);   // /log2(8)
}

// CPB MLP: table[729,2] -> 512 relu -> 24 ; one block per table row
__global__ void cpb_mlp(const float* __restrict__ w1, const float* __restrict__ b1,
                        const float* __restrict__ w2, const float* __restrict__ b2,
                        float* __restrict__ bt) {
  __shared__ float h1[512];
  int row = blockIdx.x, tid = threadIdx.x;
  float t0 = cpb_coord(row / (2 * RES - 1));
  float t1 = cpb_coord(row % (2 * RES - 1));
  for (int j = tid; j < 512; j += 256)
    h1[j] = fmaxf(0.f, t0 * w1[j] + t1 * w1[512 + j] + b1[j]);
  __syncthreads();
  if (tid < NHEAD) {
    float acc = b2[tid];
    for (int i = 0; i < 512; ++i) acc += h1[i] * w2[i * NHEAD + tid];
    bt[row * NHEAD + tid] = acc;
  }
}

// Gather + 16*sigmoid -> pre-masked bias in WMMA-C-fragment order:
// biasP[h][mt][n (224)][lm (16)]; invalid (m>=196 | n>=196) slots = -1e30.
__global__ void cpb_gather(const float* __restrict__ bt, float* __restrict__ biasP) {
  int flat = blockIdx.x * 256 + threadIdx.x;
  const int total = NHEAD * MTILES * NPAD * 16;
  if (flat >= total) return;
  int lm = flat & 15;
  int n  = (flat >> 4) % NPAD;
  int mt = ((flat >> 4) / NPAD) % MTILES;
  int h  = flat / (MTILES * NPAD * 16);
  int m = mt * 16 + lm;
  float out = NEGBIG;
  if (m < NTOK && n < NTOK) {
    int dr = m / RES - n / RES + (RES - 1);
    int dc = m % RES - n % RES + (RES - 1);
    float v = bt[(dr * (2 * RES - 1) + dc) * NHEAD + h];
    out = 16.f / (1.f + __expf(-v));
  }
  biasP[flat] = out;
}

// ---------- WMMA GEMM: C[M,N] = A[M,K] * B[K,N] + bias ----------
// block tile 128M x 128N, 8 waves (4M x 2N), wave tile 32M x 64N,
// K-step 32, software-pipelined global->reg->LDS staging.
// Per K-step per wave: 12 ds_read_b128 feed 8 WMMAs (1.5 reads/WMMA).
template <typename OutT>
__global__ __launch_bounds__(256) void gemm_f16(const _Float16* __restrict__ A,
                                                const _Float16* __restrict__ B,
                                                const float* __restrict__ bias,
                                                OutT* __restrict__ C,
                                                int M, int N, int K) {
  __shared__ _Float16 As[128][40];   // [m][k], padded rows
  __shared__ _Float16 Bt[128][40];   // [n][k], transposed, padded

  int tid  = threadIdx.x;
  int lane = tid & 31, w = tid >> 5;
  int hi = lane >> 4, l16 = lane & 15;
  int koff = hi << 3;                // A-frag k offset (halves)
  int kbB  = hi << 4;                // B-frag k base (halves)
  int m0 = (w & 3) << 5;             // 32-row wave tile
  int nb = (w >> 2) << 6;            // 64-col wave tile
  long gm0 = (long)blockIdx.x * 128;
  long gn0 = (long)blockIdx.y * 128;

  v8f acc[2][4];
#pragma unroll
  for (int s = 0; s < 2; ++s)
#pragma unroll
    for (int i = 0; i < 4; ++i) acc[s][i] = 0.f;

  int arow = tid >> 1, acg = (tid & 1) << 4;   // A stage: 16 halves/thread
  int bk = tid >> 3, bng = (tid & 7) << 4;     // B stage: 16 halves/thread
  const _Float16* Aptr = A + (gm0 + arow) * (long)K + acg;
  const _Float16* Bptr = B + (long)bk * N + gn0 + bng;

  // prologue: fetch first K-step into registers
  v4u aReg0 = *(const v4u*)(Aptr);
  v4u aReg1 = *(const v4u*)(Aptr + 8);
  H16 bReg;
  bReg.u[0] = *(const v4u*)(Bptr);
  bReg.u[1] = *(const v4u*)(Bptr + 8);

  for (int k0 = 0; k0 < K; k0 += 32) {
    *(v4u*)&As[arow][acg] = aReg0;
    *(v4u*)&As[arow][acg + 8] = aReg1;
#pragma unroll
    for (int j = 0; j < 16; ++j) Bt[bng + j][bk] = bReg.h[j];
    __syncthreads();

    // prefetch next K-step while WMMAs run
    if (k0 + 32 < K) {
      aReg0 = *(const v4u*)(Aptr + k0 + 32);
      aReg1 = *(const v4u*)(Aptr + k0 + 40);
      bReg.u[0] = *(const v4u*)(Bptr + (long)(k0 + 32) * N);
      bReg.u[1] = *(const v4u*)(Bptr + (long)(k0 + 32) * N + 8);
    }

    Frag16 a[2];
#pragma unroll
    for (int s = 0; s < 2; ++s) {
      a[s].u[0] = *(const v4u*)&As[m0 + s * 16 + l16][koff];
      a[s].u[1] = *(const v4u*)&As[m0 + s * 16 + l16][16 + koff];
    }
#pragma unroll
    for (int i = 0; i < 4; ++i) {
      Frag16 b;
      b.u[0] = *(const v4u*)&Bt[nb + i * 16 + l16][koff];
      b.u[1] = *(const v4u*)&Bt[nb + i * 16 + l16][16 + koff];
#pragma unroll
      for (int s = 0; s < 2; ++s)
        acc[s][i] = __builtin_amdgcn_wmma_f32_16x16x32_f16(
            false, a[s].h, false, b.h, (short)0, acc[s][i], false, false);
    }
    __syncthreads();
  }

#pragma unroll
  for (int i = 0; i < 4; ++i) {
    long gn = gn0 + nb + i * 16 + l16;
    float bb = bias[gn];
#pragma unroll
    for (int s = 0; s < 2; ++s) {
#pragma unroll
      for (int r = 0; r < 8; ++r) {
        long gm = gm0 + m0 + s * 16 + r + (hi << 3);
        C[gm * (long)N + gn] = (OutT)(acc[s][i][r] + bb);
      }
    }
  }
}

// ---------- Attention: one block (4 waves, 128 thr) per (b, h) ----------
// Fully branch-free: q pre-scaled by 1/sqrt(32) in the QKV GEMM; pre-masked
// bias (with -1e30 in invalid slots) enters as the WMMA C operand; OOB token
// indices are clamped so all loads execute unconditionally.
__global__ __launch_bounds__(128) void attn_wmma(const _Float16* __restrict__ qkv,
                                                 const float* __restrict__ biasP,
                                                 _Float16* __restrict__ attnOut) {
  __shared__ _Float16 vt[HDIM][NPAD + 8];   // V transposed [d][tok]
  __shared__ float    S[16][NPAD];          // score block (one 16-row q tile)
  __shared__ _Float16 P[16][NPAD + 8];      // exp probabilities (f16)
  __shared__ float    Opart[4][16][16];     // per-wave PV partials
  __shared__ float    rsum[16];

  int b = blockIdx.x / NHEAD, h = blockIdx.x % NHEAD;
  int tid = threadIdx.x, lane = tid & 31, w = tid >> 5;
  int hi = lane >> 4, l16 = lane & 15;
  int koff = hi << 3;       // A-frag k offset
  int kbB  = hi << 4;       // B-frag k base

  const _Float16* qbase = qkv + ((long)b * NTOK) * (3 * DIM) + h * HDIM;
  const _Float16* kbase = qbase + DIM;
  const _Float16* vbase = qbase + 2 * DIM;

  // stage V transposed; clamp tok (P[:,tok>=196]==0 for valid rows anyway)
  for (int i = tid; i < NPAD * (HDIM / 4); i += 128) {
    int tok = i >> 3, cg = (i & 7) << 2;
    int tc = min(tok, NTOK - 1);
    H4 hv;
    hv.u = *(const v2u*)(vbase + (long)tc * (3 * DIM) + cg);
#pragma unroll
    for (int j = 0; j < 4; ++j) vt[cg + j][tok] = hv.h[j];
  }
  __syncthreads();

  for (int mt = 0; mt < MTILES; ++mt) {
    // ---- Phase 1: S block = q' @ k^T + biasP  (bias as WMMA C operand) ----
    Frag16 aq;
    int qm = mt * 16 + l16;
    const _Float16* qp = qbase + (long)min(qm, NTOK - 1) * (3 * DIM);
    aq.u[0] = *(const v4u*)(qp + koff);
    aq.u[1] = *(const v4u*)(qp + 16 + koff);

    const float* bmt = biasP + (((long)h * MTILES + mt) * NPAD) * 16 + (hi << 3);

    for (int nt = w; nt < 14; nt += 4) {
      Frag16 bk;
      int ktok = nt * 16 + l16;
      const _Float16* kp = kbase + (long)min(ktok, NTOK - 1) * (3 * DIM);
      bk.u[0] = *(const v4u*)(kp + koff);
      bk.u[1] = *(const v4u*)(kp + 16 + koff);

      int n = nt * 16 + l16;
      F8 c;
      c.q[0] = *(const v4f*)(bmt + (long)n * 16);
      c.q[1] = *(const v4f*)(bmt + (long)n * 16 + 4);

      v8f s = __builtin_amdgcn_wmma_f32_16x16x32_f16(
          false, aq.h, false, bk.h, (short)0, c.v, false, false);
#pragma unroll
      for (int r = 0; r < 8; ++r) S[r + (hi << 3)][n] = s[r];
    }
    __syncthreads();

    // ---- Phase 2: softmax (16 rows x 8 threads/row), write f16 P ----
    {
      int srow = tid >> 3, j0 = tid & 7;
      float mx = NEGBIG;
      for (int c = j0 * 28; c < j0 * 28 + 28; ++c) mx = fmaxf(mx, S[srow][c]);
#pragma unroll
      for (int off = 1; off < 8; off <<= 1) mx = fmaxf(mx, __shfl_xor(mx, off, 32));
      float sum = 0.f;
      for (int c = j0 * 28; c < j0 * 28 + 28; ++c) {
        float e = __expf(S[srow][c] - mx);
        P[srow][c] = (_Float16)e;
        sum += e;
      }
#pragma unroll
      for (int off = 1; off < 8; off <<= 1) sum += __shfl_xor(sum, off, 32);
      if (j0 == 0) rsum[srow] = sum;
    }
    __syncthreads();

    // ---- Phase 3: O partials = P @ V (even waves d-tile 0, odd d-tile 1) ----
    {
      v8f oacc = 0.f;
      int dt = w & 1;
      for (int kt = (w >> 1); kt < 7; kt += 2) {
        Frag16 ap, bv;
        ap.u[0] = *(const v4u*)&P[l16][kt * 32 + koff];
        ap.u[1] = *(const v4u*)&P[l16][kt * 32 + 16 + koff];
        int d = dt * 16 + l16;
        bv.u[0] = *(const v4u*)&vt[d][kt * 32 + koff];
        bv.u[1] = *(const v4u*)&vt[d][kt * 32 + 16 + koff];
        oacc = __builtin_amdgcn_wmma_f32_16x16x32_f16(
            false, ap.h, false, bv.h, (short)0, oacc, false, false);
      }
#pragma unroll
      for (int r = 0; r < 8; ++r) Opart[w][r + (hi << 3)][l16] = oacc[r];
    }
    __syncthreads();

    // ---- Phase 4: reduce partials, normalize, store f16 ----
    for (int e = tid; e < 16 * HDIM; e += 128) {
      int m = e >> 5, d = e & 31;
      int gm = mt * 16 + m;
      if (gm < NTOK) {
        int dt = d >> 4, nn = d & 15;
        float val = (Opart[dt][m][nn] + Opart[dt + 2][m][nn]) / rsum[m];
        attnOut[((long)b * NTOK + gm) * DIM + h * HDIM + d] = (_Float16)val;
      }
    }
    __syncthreads();
  }
}

// ---------- launch ----------
extern "C" void kernel_launch(void* const* d_in, const int* in_sizes, int n_in,
                              void* d_out, int out_size, void* d_ws, size_t ws_size,
                              hipStream_t stream) {
  const float* x      = (const float*)d_in[0];
  const float* qkv_w  = (const float*)d_in[1];
  const float* qkv_b  = (const float*)d_in[2];
  const float* proj_w = (const float*)d_in[3];
  const float* proj_b = (const float*)d_in[4];
  const float* cpb_w1 = (const float*)d_in[5];
  const float* cpb_b1 = (const float*)d_in[6];
  const float* cpb_w2 = (const float*)d_in[7];
  const float* cpb_b2 = (const float*)d_in[8];

  const long nx  = (long)MROWS * DIM;        // 19,267,584
  const long nwq = (long)DIM * 3 * DIM;      //  1,769,472
  const long nwp = (long)DIM * DIM;          //    589,824
  const long nqkv = (long)MROWS * 3 * DIM;   // 57,802,752
  const int  nbiasP = NHEAD * MTILES * NPAD * 16;   // 1,118,208

  _Float16* x16   = (_Float16*)d_ws;
  _Float16* wq16  = x16 + nx;
  _Float16* wp16  = wq16 + nwq;
  _Float16* qkv16 = wp16 + nwp;
  _Float16* att16 = qkv16 + nqkv;
  float*    bt      = (float*)(att16 + nx);
  float*    qkv_bs  = bt + (2 * RES - 1) * (2 * RES - 1) * NHEAD;
  float*    biasP   = qkv_bs + 3 * DIM;

  // fp32 -> f16 conversions (Q columns of qkv_w pre-scaled by 1/sqrt(32))
  cvt_f16<<<(unsigned)((nx / 4 + 255) / 256), 256, 0, stream>>>(x, x16, (int)(nx / 4));
  cvt_f16_qkvw<<<(unsigned)((nwq / 4 + 255) / 256), 256, 0, stream>>>(qkv_w, wq16, (int)(nwq / 4));
  cvt_f16<<<(unsigned)((nwp / 4 + 255) / 256), 256, 0, stream>>>(proj_w, wp16, (int)(nwp / 4));
  scale_qkv_b<<<(3 * DIM + 255) / 256, 256, 0, stream>>>(qkv_b, qkv_bs);

  // continuous relative position bias -> pre-masked, C-fragment-ordered table
  cpb_mlp<<<(2 * RES - 1) * (2 * RES - 1), 256, 0, stream>>>(cpb_w1, cpb_b1, cpb_w2, cpb_b2, bt);
  cpb_gather<<<(nbiasP + 255) / 256, 256, 0, stream>>>(bt, biasP);

  // QKV projection: [25088,768] x [768,2304]
  gemm_f16<_Float16><<<dim3(MROWS / 128, (3 * DIM) / 128), 256, 0, stream>>>(
      x16, wq16, qkv_bs, qkv16, MROWS, 3 * DIM, DIM);

  // windowed attention per (b, h)
  attn_wmma<<<128 * NHEAD, 128, 0, stream>>>(qkv16, biasP, att16);

  // output projection: [25088,768] x [768,768] -> fp32 out
  gemm_f16<float><<<dim3(MROWS / 128, DIM / 128), 256, 0, stream>>>(
      att16, wp16, proj_b, (float*)d_out, MROWS, DIM, DIM);
}
